// HunterObjectHead_56779467653775
// MI455X (gfx1250) — matrix operand order repaired and checked by reading, not compiled, not run on hardware
//
#include <hip/hip_runtime.h>
#include <hip/hip_bf16.h>

// ---------------------------------------------------------------------------
// CDNA5 (gfx1250) implementation of the HunterObjectHead reference.
// wave32, WMMA f16->f32 for the two 128-wide GEMM layers; segmented
// reductions via global atomics; weights pre-packed into WMMA B-fragment
// layout (f16). Shape-MLP B fragments staged in LDS; atomics/stores use
// dump-row redirection instead of divergent guards (EXEC stays full).
// ---------------------------------------------------------------------------

typedef __attribute__((ext_vector_type(16))) _Float16 v16h;
typedef __attribute__((ext_vector_type(8)))  float    v8f;

#define WMMA_F16(a, b, c) \
  __builtin_amdgcn_wmma_f32_16x16x32_f16(false, (a), false, (b), (short)0, (c), false, false)

// 16-bit A-matrix 16x32 layout: lane (m = l&15, half = l>>4) element t holds
// K = 8*half + t          (t in 0..7)
// K = 16 + 8*half + (t-8) (t in 8..15)
__device__ __forceinline__ int kpatA(int t, int half) {
  return (t < 8) ? (8 * half + t) : (16 + 8 * half + (t - 8));
}

// Generic float atomic max (works for mixed-sign values; init with -FLT_MAX).
__device__ __forceinline__ void atomicMaxF(float* addr, float v) {
  if (v >= 0.0f) atomicMax(reinterpret_cast<int*>(addr), __float_as_int(v));
  else           atomicMin(reinterpret_cast<unsigned int*>(addr), __float_as_uint(v));
}

// ---------------------------------------------------------------------------
// small utility kernels
// ---------------------------------------------------------------------------
__global__ void k_fill_f32(float* __restrict__ p, long n, float v) {
  long i = (long)blockIdx.x * blockDim.x + threadIdx.x;
  if (i < n) p[i] = v;
}

// Pack a (K x 128) f32 weight into f16 WMMA B-fragment-major layout:
// dst[((kc*8+nt)*32 + lane)*16 + t] = W[(kc*32 + kpatA(t, lane>>4))*128 + nt*16 + (lane&15)]
__device__ __forceinline__ void pack128(const float* __restrict__ W,
                                        _Float16* __restrict__ dst, int idx) {
  int t  = idx & 15;
  int l  = (idx >> 4) & 31;
  int nt = (idx >> 9) & 7;
  int kc = idx >> 12;
  int k  = kc * 32 + kpatA(t, l >> 4);
  int n  = (nt << 4) + (l & 15);
  dst[idx] = (_Float16)W[k * 128 + n];
}

__global__ void k_wprep(const float* __restrict__ W2s, const float* __restrict__ W1l,
                        const float* __restrict__ W2l,
                        _Float16* __restrict__ f2s, _Float16* __restrict__ f1l,
                        _Float16* __restrict__ f2l) {
  int idx = blockIdx.x * blockDim.x + threadIdx.x;           // 0 .. 65535
  if (idx < 16384)        pack128(W2s, f2s, idx);            // 128x128
  else if (idx < 49152)   pack128(W1l, f1l, idx - 16384);    // first 256 rows of (262x128)
  else if (idx < 65536)   pack128(W2l, f2l, idx - 49152);    // 128x128
}

__global__ void k_cent_accum(const float* __restrict__ fg_xyz,
                             const int* __restrict__ locals2fg,
                             float* __restrict__ cent_sum, float* __restrict__ cnt_l,
                             int N) {
  int i = blockIdx.x * blockDim.x + threadIdx.x;
  if (i >= N) return;
  int lid = locals2fg[i];
  atomicAdd(&cent_sum[lid * 3 + 0], fg_xyz[i * 3 + 0]);
  atomicAdd(&cent_sum[lid * 3 + 1], fg_xyz[i * 3 + 1]);
  atomicAdd(&cent_sum[lid * 3 + 2], fg_xyz[i * 3 + 2]);
  atomicAdd(&cnt_l[lid], 1.0f);
}

__global__ void k_cnti(const int* __restrict__ inst2locals, float* __restrict__ cnt_i, int L) {
  int l = blockIdx.x * blockDim.x + threadIdx.x;
  if (l < L) atomicAdd(&cnt_i[inst2locals[l]], 1.0f);
}

__global__ void k_centroid(const float* __restrict__ cent_sum, const float* __restrict__ cnt_l,
                           float* __restrict__ centroid, int L) {
  int l = blockIdx.x * blockDim.x + threadIdx.x;
  if (l >= L) return;
  float inv = 1.0f / fmaxf(cnt_l[l], 1.0f);
  centroid[l * 3 + 0] = cent_sum[l * 3 + 0] * inv;
  centroid[l * 3 + 1] = cent_sum[l * 3 + 1] * inv;
  centroid[l * 3 + 2] = cent_sum[l * 3 + 2] * inv;
}

__global__ void k_gtarget(const int* __restrict__ idx_sweep, const float* __restrict__ centroid,
                          float* __restrict__ gtarget, int I) {
  int i = blockIdx.x * blockDim.x + threadIdx.x;
  if (i >= I) return;
  int li = idx_sweep[i];
  gtarget[i * 3 + 0] = centroid[li * 3 + 0];
  gtarget[i * 3 + 1] = centroid[li * 3 + 1];
  gtarget[i * 3 + 2] = centroid[li * 3 + 2];
}

// segment-max of fg_feat (N,128) over locals2fg. One thread per float4.
// Pure 512MB stream: prefetch ahead (global_prefetch_b8) to hide atomic latency.
__global__ void k_featmax(const float* __restrict__ fg_feat, const int* __restrict__ locals2fg,
                          float* __restrict__ feat_max, int N) {
  int idx = blockIdx.x * blockDim.x + threadIdx.x;  // N*32
  if (idx >= N * 32) return;
  const float4* f4 = reinterpret_cast<const float4*>(fg_feat);
  __builtin_prefetch(f4 + idx + 8192, 0, 0);        // speculative; OOB dropped by HW
  int i = idx >> 5;
  int g = idx & 31;
  const float4 v = f4[idx];
  int lid = locals2fg[i];
  float* base = feat_max + (long)lid * 128 + (g << 2);
  atomicMaxF(base + 0, v.x);
  atomicMaxF(base + 1, v.y);
  atomicMaxF(base + 2, v.z);
  atomicMaxF(base + 3, v.w);
}

// locals_feat = shape_max + (present ? feat_max : 0) ; then atomic-max into globals
__global__ void k_locfeat(float* __restrict__ shape_max /*in/out -> locals_feat*/,
                          const float* __restrict__ feat_max,
                          const float* __restrict__ cnt_l,
                          const int* __restrict__ inst2locals,
                          float* __restrict__ gfeat, int L) {
  int idx = blockIdx.x * blockDim.x + threadIdx.x;  // L*128
  if (idx >= L * 128) return;
  int l = idx >> 7;
  int c = idx & 127;
  float fm = (cnt_l[l] > 0.0f) ? feat_max[idx] : 0.0f;
  float v = shape_max[idx] + fm;
  shape_max[idx] = v;  // becomes locals_feat
  atomicMaxF(&gfeat[(long)inst2locals[l] * 128 + c], v);
}

__global__ void k_globfin(float* __restrict__ gfeat, const float* __restrict__ cnt_i, int I) {
  int idx = blockIdx.x * blockDim.x + threadIdx.x;  // I*128
  if (idx >= I * 128) return;
  int i = idx >> 7;
  if (!(cnt_i[i] > 0.0f)) gfeat[idx] = 0.0f;
}

// ---------------------------------------------------------------------------
// shape MLP: per 16-point tile per wave. layer1 (3->128) in f32 VALU directly
// into A-fragment layout; layer2 (128->128) via 32x v_wmma_f32_16x16x32_f16
// fed from LDS-staged B fragments; relu + int-bits atomicMax into shape_max
// (values >= 0, init 0). Invalid tail rows redirect to dump row L (no branch).
// ---------------------------------------------------------------------------
__global__ __launch_bounds__(256)
void k_shape_mlp(const float* __restrict__ fg_xyz,
                 const int* __restrict__ locals2fg,
                 const float* __restrict__ centroid,
                 const float* __restrict__ W1, const float* __restrict__ b1,
                 const _Float16* __restrict__ W2frag, const float* __restrict__ b2,
                 float* __restrict__ shape_max, int N, int Ldump) {
  __shared__ float    sW1[384];
  __shared__ float    sb1[128];
  __shared__ float    sb2[128];
  __shared__ _Float16 sB[16384];          // 32KB: packed W2 fragments
  for (int i = threadIdx.x; i < 384; i += 256) sW1[i] = W1[i];
  for (int i = threadIdx.x; i < 128; i += 256) { sb1[i] = b1[i]; sb2[i] = b2[i]; }
  {
    const uint4* src = reinterpret_cast<const uint4*>(W2frag);
    uint4* dst = reinterpret_cast<uint4*>(sB);
    for (int i = threadIdx.x; i < 2048; i += 256) dst[i] = src[i];
  }
  __syncthreads();

  const int wave = threadIdx.x >> 5;
  const int lane = threadIdx.x & 31;
  const int m    = lane & 15;
  const int half = lane >> 4;
  const int tile = blockIdx.x * 8 + wave;
  const int base = tile * 16;

  int rowA = base + m; if (rowA > N - 1) rowA = N - 1;   // clamp loads
  int lidA = locals2fg[rowA];
  float x0 = fg_xyz[rowA * 3 + 0] - centroid[lidA * 3 + 0];
  float x1 = fg_xyz[rowA * 3 + 1] - centroid[lidA * 3 + 1];
  float x2 = fg_xyz[rowA * 3 + 2] - centroid[lidA * 3 + 2];

  v16h a[4];
#pragma unroll
  for (int kc = 0; kc < 4; ++kc)
#pragma unroll
    for (int t = 0; t < 16; ++t) {
      int j = kc * 32 + kpatA(t, half);
      float hv = fmaf(x0, sW1[j], fmaf(x1, sW1[128 + j], fmaf(x2, sW1[256 + j], sb1[j])));
      a[kc][t] = (_Float16)fmaxf(hv, 0.0f);
    }

  int lids[8];
#pragma unroll
  for (int r = 0; r < 8; ++r) {
    int rr = base + r + 8 * half;
    // invalid tail rows -> dump row Ldump (atomic is harmless there)
    lids[r] = (rr < N) ? locals2fg[rr] : Ldump;
  }
  const int ncol = lane & 15;
#pragma unroll
  for (int nt = 0; nt < 8; ++nt) {
    v8f c = {0.f, 0.f, 0.f, 0.f, 0.f, 0.f, 0.f, 0.f};
#pragma unroll
    for (int kc = 0; kc < 4; ++kc) {
      v16h b = *reinterpret_cast<const v16h*>(sB + ((((kc * 8 + nt) * 32) + lane) << 4));
      c = WMMA_F16(a[kc], b, c);
    }
    int col = nt * 16 + ncol;
    float bias = sb2[col];
#pragma unroll
    for (int r = 0; r < 8; ++r) {
      float v = fmaxf(c[r] + bias, 0.0f);  // relu; >= 0 so int atomicMax is valid
      atomicMax(reinterpret_cast<int*>(shape_max + (long)lids[r] * 128 + col),
                __float_as_int(v));
    }
  }
}

// ---------------------------------------------------------------------------
// local MLP: 262->128->128. K=256 part via WMMA (8 chunks), 6-dim geometry
// tail + bias folded into the accumulator init. Inter-layer tile bounced via
// per-wave LDS (LDS ops are in-order within a wave). Tail-row stores redirect
// to a dump buffer via pointer select (no divergence).
// ---------------------------------------------------------------------------
__global__ __launch_bounds__(128)
void k_loc_mlp(const float* __restrict__ lf,        // locals_feat (L,128)
               const float* __restrict__ gf,        // globals_feat (I,128)
               const float* __restrict__ centroid,  // (L,3)
               const float* __restrict__ gtarget,   // (I,3)
               const int* __restrict__ inst2locals,
               const float* __restrict__ W1g,       // full W1_loc (262,128) f32
               const float* __restrict__ b1,
               const _Float16* __restrict__ W1frag, // packed rows 0..255
               const _Float16* __restrict__ W2frag, const float* __restrict__ b2,
               float* __restrict__ feat_out,        // (L,128)
               float* __restrict__ dump,            // >=128 floats scratch
               int L) {
  __shared__ _Float16 sh[4][16][136];
  const int wave = threadIdx.x >> 5;
  const int lane = threadIdx.x & 31;
  const int m    = lane & 15;
  const int half = lane >> 4;
  const int tile = blockIdx.x * 4 + wave;
  const int base = tile * 16;

  int rowA = base + m; if (rowA > L - 1) rowA = L - 1;
  int instA = inst2locals[rowA];
  v16h a1[8];
#pragma unroll
  for (int kc = 0; kc < 8; ++kc)
#pragma unroll
    for (int t = 0; t < 16; ++t) {
      int j = kc * 32 + kpatA(t, half);
      float v = (j < 128) ? lf[(long)rowA * 128 + j] : gf[(long)instA * 128 + (j - 128)];
      a1[kc][t] = (_Float16)v;
    }

  int rows[8];
  float geo[8][6];
#pragma unroll
  for (int r = 0; r < 8; ++r) {
    int rr = base + r + 8 * half;
    rows[r] = rr;
    int rc = rr < L ? rr : L - 1;
    int ins = inst2locals[rc];
    geo[r][0] = centroid[rc * 3 + 0];
    geo[r][1] = centroid[rc * 3 + 1];
    geo[r][2] = centroid[rc * 3 + 2];
    geo[r][3] = gtarget[ins * 3 + 0];
    geo[r][4] = gtarget[ins * 3 + 1];
    geo[r][5] = gtarget[ins * 3 + 2];
  }

  const int ncol = lane & 15;
  // ---- layer 1 ----
#pragma unroll
  for (int nt = 0; nt < 8; ++nt) {
    int col = nt * 16 + ncol;
    float wg[6];
#pragma unroll
    for (int k = 0; k < 6; ++k) wg[k] = W1g[(256 + k) * 128 + col];
    float bb = b1[col];
    v8f c;
#pragma unroll
    for (int r = 0; r < 8; ++r) {
      float acc = bb;
#pragma unroll
      for (int k = 0; k < 6; ++k) acc = fmaf(geo[r][k], wg[k], acc);
      c[r] = acc;
    }
#pragma unroll
    for (int kc = 0; kc < 8; ++kc) {
      v16h b = *reinterpret_cast<const v16h*>(W1frag + ((((kc * 8 + nt) * 32) + lane) << 4));
      c = WMMA_F16(a1[kc], b, c);
    }
#pragma unroll
    for (int r = 0; r < 8; ++r)
      sh[wave][r + 8 * half][col] = (_Float16)fmaxf(c[r], 0.0f);
  }

  // ---- re-enter A layout from LDS ----
  v16h a2[4];
#pragma unroll
  for (int kc = 0; kc < 4; ++kc)
#pragma unroll
    for (int t = 0; t < 16; ++t)
      a2[kc][t] = sh[wave][m][kc * 32 + kpatA(t, half)];

  // ---- layer 2 ----
#pragma unroll
  for (int nt = 0; nt < 8; ++nt) {
    v8f c = {0.f, 0.f, 0.f, 0.f, 0.f, 0.f, 0.f, 0.f};
#pragma unroll
    for (int kc = 0; kc < 4; ++kc) {
      v16h b = *reinterpret_cast<const v16h*>(W2frag + ((((kc * 8 + nt) * 32) + lane) << 4));
      c = WMMA_F16(a2[kc], b, c);
    }
    int col = nt * 16 + ncol;
    float bias = b2[col];
#pragma unroll
    for (int r = 0; r < 8; ++r) {
      float* dst = (rows[r] < L) ? (feat_out + (long)rows[r] * 128 + col) : (dump + col);
      *dst = fmaxf(c[r] + bias, 0.0f);
    }
  }
}

// decoder: (L,128) @ (128,7) + b
__global__ __launch_bounds__(256)
void k_dec(const float* __restrict__ feat, const float* __restrict__ Wd,
           const float* __restrict__ bd, float* __restrict__ out, int L) {
  __shared__ float sWd[128 * 7];
  __shared__ float sbd[7];
  for (int i = threadIdx.x; i < 896; i += 256) sWd[i] = Wd[i];
  if (threadIdx.x < 7) sbd[threadIdx.x] = bd[threadIdx.x];
  __syncthreads();
  int row = blockIdx.x * 256 + threadIdx.x;
  if (row >= L) return;
  float acc[7];
#pragma unroll
  for (int j = 0; j < 7; ++j) acc[j] = sbd[j];
  const float* f = feat + (long)row * 128;
  for (int k = 0; k < 128; ++k) {
    float fv = f[k];
#pragma unroll
    for (int j = 0; j < 7; ++j) acc[j] = fmaf(fv, sWd[k * 7 + j], acc[j]);
  }
#pragma unroll
  for (int j = 0; j < 7; ++j) out[(long)row * 7 + j] = acc[j];
}

// ---------------------------------------------------------------------------
extern "C" void kernel_launch(void* const* d_in, const int* in_sizes, int n_in,
                              void* d_out, int out_size, void* d_ws, size_t ws_size,
                              hipStream_t stream) {
  const float* fg_xyz      = (const float*)d_in[0];
  const float* fg_feat     = (const float*)d_in[1];
  const float* W1s         = (const float*)d_in[2];
  const float* b1s         = (const float*)d_in[3];
  const float* W2s         = (const float*)d_in[4];
  const float* b2s         = (const float*)d_in[5];
  const float* W1l         = (const float*)d_in[6];
  const float* b1l         = (const float*)d_in[7];
  const float* W2l         = (const float*)d_in[8];
  const float* b2l         = (const float*)d_in[9];
  const float* Wd          = (const float*)d_in[10];
  const float* bd          = (const float*)d_in[11];
  const int*   locals2fg   = (const int*)d_in[12];
  const int*   inst2locals = (const int*)d_in[13];
  const int*   idx_sweep   = (const int*)d_in[14];

  const int N = in_sizes[12];   // locals2fg length
  const int L = in_sizes[13];   // inst2locals length
  const int I = in_sizes[14];   // indices_locals_max_sweep length

  // ---- carve workspace (256B-aligned regions) ----
  char* p = (char*)d_ws;
  auto carve = [&p](size_t bytes) -> char* {
    char* r = p;
    p += (bytes + 255) & ~(size_t)255;
    return r;
  };
  float*    cent_sum  = (float*)carve((size_t)L * 3 * 4);
  float*    cnt_l     = (float*)carve((size_t)L * 4);
  float*    centroid  = (float*)carve((size_t)L * 3 * 4);
  float*    shape_max = (float*)carve((size_t)(L + 1) * 128 * 4);  // +1 dump row
  float*    feat_max  = (float*)carve((size_t)L * 128 * 4);
  float*    cnt_i     = (float*)carve((size_t)I * 4);
  float*    gfeat     = (float*)carve((size_t)I * 128 * 4);
  float*    gtarget   = (float*)carve((size_t)I * 3 * 4);
  float*    dump      = (float*)carve(256 * 4);
  _Float16* W2s_frag  = (_Float16*)carve((size_t)128 * 128 * 2);
  _Float16* W1l_frag  = (_Float16*)carve((size_t)256 * 128 * 2);
  _Float16* W2l_frag  = (_Float16*)carve((size_t)128 * 128 * 2);

  float* out_tf   = (float*)d_out;                 // (L,7)
  float* feat_out = (float*)d_out + (size_t)L * 7; // (L,128)

  const float NEG = -3.402823466e38f;
  auto blocks = [](long n, int bs) { return (unsigned)((n + bs - 1) / bs); };

  // ---- init accumulators ----
  k_fill_f32<<<blocks((long)L * 3, 256), 256, 0, stream>>>(cent_sum, (long)L * 3, 0.0f);
  k_fill_f32<<<blocks(L, 256), 256, 0, stream>>>(cnt_l, L, 0.0f);
  k_fill_f32<<<blocks((long)(L + 1) * 128, 256), 256, 0, stream>>>(shape_max,
                                                                   (long)(L + 1) * 128, 0.0f);
  k_fill_f32<<<blocks((long)L * 128, 256), 256, 0, stream>>>(feat_max, (long)L * 128, NEG);
  k_fill_f32<<<blocks(I, 256), 256, 0, stream>>>(cnt_i, I, 0.0f);
  k_fill_f32<<<blocks((long)I * 128, 256), 256, 0, stream>>>(gfeat, (long)I * 128, NEG);

  // ---- weight packing (f32 -> f16 fragment-major) ----
  k_wprep<<<blocks(65536, 256), 256, 0, stream>>>(W2s, W1l, W2l, W2s_frag, W1l_frag, W2l_frag);

  // ---- centroids ----
  k_cent_accum<<<blocks(N, 256), 256, 0, stream>>>(fg_xyz, locals2fg, cent_sum, cnt_l, N);
  k_cnti<<<blocks(L, 256), 256, 0, stream>>>(inst2locals, cnt_i, L);
  k_centroid<<<blocks(L, 256), 256, 0, stream>>>(cent_sum, cnt_l, centroid, L);
  k_gtarget<<<blocks(I, 256), 256, 0, stream>>>(idx_sweep, centroid, gtarget, I);

  // ---- shape MLP + segment-max (WMMA) ----
  {
    int tiles = (N + 15) / 16;
    k_shape_mlp<<<blocks(tiles, 8), 256, 0, stream>>>(fg_xyz, locals2fg, centroid,
                                                      W1s, b1s, W2s_frag, b2s,
                                                      shape_max, N, L /*dump row*/);
  }
  // ---- fg_feat segment-max ----
  k_featmax<<<blocks((long)N * 32, 256), 256, 0, stream>>>(fg_feat, locals2fg, feat_max, N);

  // ---- combine locals_feat + globals segmax ----
  k_locfeat<<<blocks((long)L * 128, 256), 256, 0, stream>>>(shape_max, feat_max, cnt_l,
                                                            inst2locals, gfeat, L);
  k_globfin<<<blocks((long)I * 128, 256), 256, 0, stream>>>(gfeat, cnt_i, I);

  // ---- local MLP (WMMA) ----
  {
    int tiles = (L + 15) / 16;
    k_loc_mlp<<<blocks(tiles, 4), 128, 0, stream>>>(shape_max /*locals_feat*/, gfeat,
                                                    centroid, gtarget, inst2locals,
                                                    W1l, b1l, W1l_frag, W2l_frag, b2l,
                                                    feat_out, dump, L);
  }
  // ---- decoder ----
  k_dec<<<blocks(L, 256), 256, 0, stream>>>(feat_out, Wd, bd, out_tf, L);
}
